// NaViT_ROPE_22385369546996
// MI455X (gfx1250) — matrix operand order repaired and checked by reading, not compiled, not run on hardware
//
#include <hip/hip_runtime.h>
#include <hip/hip_bf16.h>
#include <math.h>
#include <stdint.h>

// ---------------- model constants ----------------
#define BATCH  4
#define NMAX   1024
#define EMBED  384
#define HEADS  6
#define HD     64
#define DEPTH  12
#define PD     96
#define FF     1536
#define QKVC   1152
#define MROWS  (BATCH * NMAX)   // 4096 token rows (padded)

typedef __attribute__((ext_vector_type(16))) _Float16 v16h;
typedef __attribute__((ext_vector_type(8)))  float    v8f;
typedef unsigned int u32x4 __attribute__((ext_vector_type(4)));
typedef int          i32x8 __attribute__((ext_vector_type(8)));
typedef int          i32x4 __attribute__((ext_vector_type(4)));

__device__ __forceinline__ v8f wmma_f16(v16h a, v16h b, v8f c) {
  // D = A(16x32 f16) * B(32x16 f16) + C(16x16 f32)
  return __builtin_amdgcn_wmma_f32_16x16x32_f16(
      /*neg_a=*/false, a, /*neg_b=*/false, b,
      /*c_mod=*/(short)0, c, /*reuse_a=*/false, /*reuse_b=*/false);
}

// async global->LDS copy of 16 bytes per lane (ASYNCcnt-tracked direct path,
// no VGPR round trip). lds = low 32 bits of generic shared pointer (LDS aperture
// keeps the offset in addr[31:0] per ISA 10.2).
__device__ __forceinline__ void async_copy_b128(unsigned lds, const void* gptr) {
  asm volatile("global_load_async_to_lds_b128 %0, %1, off"
               :: "v"(lds), "v"(gptr) : "memory");
}
__device__ __forceinline__ void wait_asynccnt0() {
  asm volatile("s_wait_asynccnt 0x0" ::: "memory");
}

// ---------------------------------------------------------------------------
// prep: patchify + LN(pd=96) -> f16 A-matrix [MROWS, 96]; axial RoPE tables.
// grid (NMAX, BATCH), block 128
// ---------------------------------------------------------------------------
__global__ void prep_kernel(const float* __restrict__ img0, const float* __restrict__ img1,
                            const float* __restrict__ img2, const float* __restrict__ img3,
                            const float* __restrict__ lnw, const float* __restrict__ lnb,
                            _Float16* __restrict__ hA96,
                            float* __restrict__ cosb, float* __restrict__ sinb) {
  __shared__ float red[128];
  const int n = blockIdx.x, b = blockIdx.y, tid = threadIdx.x;
  const int lens_[4] = {512, 768, 896, 1024};
  const int pws_[4]  = {128, 192, 224, 256};
  const float* imgs[4] = {img0, img1, img2, img3};
  const int len = lens_[b], pw = pws_[b], W = pw * 4;
  const bool valid = n < len;
  const int tx = n % pw, ty = n / pw;

  float v = 0.f;
  if (valid && tid < PD) {
    int c = tid / 32, rem = tid % 32, py = rem / 4, px = rem % 4;
    v = imgs[b][c * 32 * W + (ty * 8 + py) * W + (tx * 4 + px)];
  }
  red[tid] = (tid < PD) ? v : 0.f;
  __syncthreads();
  for (int s = 64; s > 0; s >>= 1) { if (tid < s) red[tid] += red[tid + s]; __syncthreads(); }
  const float mean = red[0] / (float)PD;
  __syncthreads();
  float d = (tid < PD) ? (v - mean) : 0.f;
  red[tid] = d * d;
  __syncthreads();
  for (int s = 64; s > 0; s >>= 1) { if (tid < s) red[tid] += red[tid + s]; __syncthreads(); }
  const float rstd = rsqrtf(red[0] / (float)PD + 1e-5f);

  if (tid < PD) {
    float o = valid ? ((v - mean) * rstd * lnw[tid] + lnb[tid]) : 0.f;
    hA96[(size_t)(b * NMAX + n) * PD + tid] = (_Float16)o;
  }
  if (tid >= 96) {                 // 32 threads: axial rope tables
    int j = tid - 96;              // 0..31 ; first 16 use tx, last 16 use ty
    int fi = j & 15;
    float f = powf(100.0f, -(float)fi / 16.0f);   // THETA^{-4i/HD}
    float ang = ((j < 16) ? (float)tx : (float)ty) * f;
    size_t o = (size_t)(b * NMAX + n) * 32 + j;
    cosb[o] = valid ? cosf(ang) : 0.f;
    sinb[o] = valid ? sinf(ang) : 0.f;
  }
}

// ---------------------------------------------------------------------------
// LayerNorm over EMBED=384: f32 in -> f16 out (GEMM A) or f32 out.
// grid MROWS, block 128 (3 elements/thread)
// ---------------------------------------------------------------------------
__global__ void ln_kernel(const float* __restrict__ x, const float* __restrict__ wv,
                          const float* __restrict__ bv,
                          _Float16* __restrict__ out16, float* __restrict__ out32) {
  __shared__ float red[128];
  const int row = blockIdx.x, tid = threadIdx.x;
  const float* r = x + (size_t)row * EMBED;
  float v[3], s = 0.f;
#pragma unroll
  for (int i = 0; i < 3; ++i) { v[i] = r[tid + i * 128]; s += v[i]; }
  red[tid] = s; __syncthreads();
  for (int st = 64; st > 0; st >>= 1) { if (tid < st) red[tid] += red[tid + st]; __syncthreads(); }
  const float mean = red[0] / (float)EMBED;
  __syncthreads();
  s = 0.f;
#pragma unroll
  for (int i = 0; i < 3; ++i) { float d = v[i] - mean; s += d * d; }
  red[tid] = s; __syncthreads();
  for (int st = 64; st > 0; st >>= 1) { if (tid < st) red[tid] += red[tid + st]; __syncthreads(); }
  const float rstd = rsqrtf(red[0] / (float)EMBED + 1e-5f);
#pragma unroll
  for (int i = 0; i < 3; ++i) {
    int c = tid + i * 128;
    float o = (v[i] - mean) * rstd * wv[c] + bv[c];
    if (out16) out16[(size_t)row * EMBED + c] = (_Float16)o;
    else       out32[(size_t)row * EMBED + c] = o;
  }
}

// ---------------------------------------------------------------------------
// Tiled WMMA GEMM: C[M,N] = A_f16[M,K] * W_f32[K,N] (+bias). 64x128 block tile,
// 256 threads = 8 waves, each wave a 16x64 output slab (4 WMMAs / K-step).
// A tile staged with GLOBAL_LOAD_ASYNC_TO_LDS_B128 (ASYNCcnt path).
// MODE 0: store f32. MODE 1: residual add into f32. MODE 2: exact GELU -> f16.
// ---------------------------------------------------------------------------
template <int MODE>
__global__ void gemm_kernel(const _Float16* __restrict__ A, const float* __restrict__ W,
                            const float* __restrict__ bias,
                            float* __restrict__ outf, _Float16* __restrict__ outh,
                            int N, int K) {
  __shared__ _Float16 As[64][32];     // 4 KB
  __shared__ _Float16 Bs[32][128];    // 8 KB
  const int tid = threadIdx.x;
  const int lane = tid & 31, wave = tid >> 5;
  const int bm = blockIdx.y * 64, bn = blockIdx.x * 128;
  const int wm = (wave >> 1) * 16;   // wave's M offset in tile
  const int wn = (wave & 1) * 64;    // wave's N offset in tile
  const int lh = lane & 15, hh = lane >> 4;
  v8f acc0 = {}, acc1 = {}, acc2 = {}, acc3 = {};

  const int rA = tid >> 2, cA = (tid & 3) * 8;
  const unsigned ldsA = (unsigned)(uintptr_t)&As[rA][cA];
  const int rB = tid >> 3, cB = (tid & 7) * 16;

  for (int k0 = 0; k0 < K; k0 += 32) {
    // A tile 64x32 halfs: direct async global->LDS (16B per lane)
    async_copy_b128(ldsA, A + (size_t)(bm + rA) * K + k0 + cA);
    { // W tile 32x128: f32 -> f16 convert into LDS (16 elems per thread)
      const float* src = W + (size_t)(k0 + rB) * N + bn + cB;
#pragma unroll
      for (int q4 = 0; q4 < 4; ++q4) {
        float4 w = *(const float4*)(src + q4 * 4);
        Bs[rB][cB + q4 * 4 + 0] = (_Float16)w.x;
        Bs[rB][cB + q4 * 4 + 1] = (_Float16)w.y;
        Bs[rB][cB + q4 * 4 + 2] = (_Float16)w.z;
        Bs[rB][cB + q4 * 4 + 3] = (_Float16)w.w;
      }
    }
    wait_asynccnt0();       // our async LDS writes complete before we signal
    __syncthreads();

    // A fragment (16x32, ISA 16-bit A layout: lanes 0-15 M, hh selects K half)
    v16h a;
#pragma unroll
    for (int p = 0; p < 8; ++p) {
      int kk = (p < 4) ? (hh * 8 + 2 * p) : (16 + hh * 8 + 2 * (p - 4));
      a[2 * p]     = As[wm + lh][kk];
      a[2 * p + 1] = As[wm + lh][kk + 1];
    }
    // B fragments (32x16: lane%16 = N column, 16 halfs = K rows in A's table order); 4 N-subtiles
    v16h b0, b1, b2, b3;
#pragma unroll
    for (int i = 0; i < 16; ++i) {
      int kr = (i < 8) ? (hh * 8 + i) : (8 + hh * 8 + i);
      b0[i] = Bs[kr][wn + lh];
      b1[i] = Bs[kr][wn + 16 + lh];
      b2[i] = Bs[kr][wn + 32 + lh];
      b3[i] = Bs[kr][wn + 48 + lh];
    }
    acc0 = wmma_f16(a, b0, acc0);
    acc1 = wmma_f16(a, b1, acc1);
    acc2 = wmma_f16(a, b2, acc2);
    acc3 = wmma_f16(a, b3, acc3);
    __syncthreads();
  }

  // epilogue: C layout — lane col = lh, VGPR r -> row r + 8*hh
#pragma unroll
  for (int r = 0; r < 8; ++r) {
    int gm = bm + wm + r + 8 * hh;
    float vv[4] = {acc0[r], acc1[r], acc2[r], acc3[r]};
#pragma unroll
    for (int t = 0; t < 4; ++t) {
      int gn = bn + wn + t * 16 + lh;
      float v0 = vv[t] + (bias ? bias[gn] : 0.f);
      size_t i0 = (size_t)gm * N + gn;
      if (MODE == 0)      outf[i0] = v0;
      else if (MODE == 1) outf[i0] += v0;
      else                outh[i0] = (_Float16)(0.5f * v0 * (1.0f + erff(v0 * 0.70710678f)));
    }
  }
}

// ---------------------------------------------------------------------------
// RoPE + head split: qkv f32 [tok,1152] -> q,k,v f16 [b,h,n,64]; q pre-scaled.
// grid (NMAX, BATCH), block 128
// ---------------------------------------------------------------------------
__global__ void rope_kernel(const float* __restrict__ qkv, const float* __restrict__ cosb,
                            const float* __restrict__ sinb,
                            _Float16* __restrict__ qf, _Float16* __restrict__ kf,
                            _Float16* __restrict__ vf) {
  const int n = blockIdx.x, b = blockIdx.y, tid = threadIdx.x;
  const float* row = qkv + (size_t)(b * NMAX + n) * QKVC;
  const float* cs = cosb + (size_t)(b * NMAX + n) * 32;
  const float* sn = sinb + (size_t)(b * NMAX + n) * 32;
  for (int i = tid; i < HEADS * 32; i += 128) {
    int head = i >> 5, p = i & 31;
    float c = cs[p], s = sn[p];
    size_t o = ((size_t)(b * HEADS + head) * NMAX + n) * HD + 2 * p;
    {
      float x0 = row[(0 * HEADS + head) * HD + 2 * p];
      float x1 = row[(0 * HEADS + head) * HD + 2 * p + 1];
      qf[o]     = (_Float16)((x0 * c - x1 * s) * 0.125f);   // * HD^-0.5
      qf[o + 1] = (_Float16)((x0 * s + x1 * c) * 0.125f);
    }
    {
      float x0 = row[(1 * HEADS + head) * HD + 2 * p];
      float x1 = row[(1 * HEADS + head) * HD + 2 * p + 1];
      kf[o]     = (_Float16)(x0 * c - x1 * s);
      kf[o + 1] = (_Float16)(x0 * s + x1 * c);
    }
  }
  for (int i = tid; i < EMBED; i += 128) {
    int head = i / HD, d = i % HD;
    vf[((size_t)(b * HEADS + head) * NMAX + n) * HD + d] =
        (_Float16)row[(2 * HEADS + head) * HD + d];
  }
}

// ---------------------------------------------------------------------------
// Flash attention, all-WMMA. grid (NMAX/128, HEADS, BATCH), 256 thr = 8 waves.
// Each wave: 16 query rows. Key chunks of 32 (lens are multiples of 32, so
// the -inf padding mask == skipping chunks beyond len).
// V chunk (contiguous 4 KB) is DMA'd by the Tensor Data Mover (wave 0 issues a
// 1-D D# descriptor, TENSORcnt-tracked); all waves transpose K into LDS.
// ---------------------------------------------------------------------------
__global__ void attn_kernel(const _Float16* __restrict__ q, const _Float16* __restrict__ k,
                            const _Float16* __restrict__ v, _Float16* __restrict__ hO,
                            int4 lens) {
  __shared__ _Float16 Kt[64][32];       // [d][key]  (transposed: B-frag lane = K-dim)
  __shared__ _Float16 Vt[32][64];       // [key][d]  (TDM destination)
  __shared__ _Float16 Ps[8][16][32];    // per-wave P staging for A-frag relayout
  const int tid = threadIdx.x, lane = tid & 31, wave = tid >> 5;
  const int h = blockIdx.y, b = blockIdx.z;
  const int lh = lane & 15, hh = lane >> 4;
  const int nkv = (b == 0) ? lens.x : (b == 1) ? lens.y : (b == 2) ? lens.z : lens.w;
  const int qrow = blockIdx.x * 128 + wave * 16;
  const _Float16* qp = q + ((size_t)(b * HEADS + h) * NMAX) * HD;
  const _Float16* kp = k + ((size_t)(b * HEADS + h) * NMAX) * HD;
  const _Float16* vp = v + ((size_t)(b * HEADS + h) * NMAX) * HD;

  // Q fragments for d=0..31 and d=32..63 (A layout)
  v16h qa0, qa1;
  {
    const _Float16* r = qp + (size_t)(qrow + lh) * HD;
#pragma unroll
    for (int p = 0; p < 8; ++p) {
      int kk = (p < 4) ? (hh * 8 + 2 * p) : (16 + hh * 8 + 2 * (p - 4));
      qa0[2 * p] = r[kk];      qa0[2 * p + 1] = r[kk + 1];
      qa1[2 * p] = r[kk + 32]; qa1[2 * p + 1] = r[kk + 33];
    }
  }
  float mrow[8], lrow[8];
  v8f O0 = {}, O1 = {}, O2 = {}, O3 = {};
#pragma unroll
  for (int r = 0; r < 8; ++r) { mrow[r] = -1e30f; lrow[r] = 0.f; }

  const unsigned ldsV = (unsigned)(uintptr_t)&Vt[0][0];

  for (int kc = 0; kc < nkv; kc += 32) {
    if (wave == 0) {
      // --- Tensor Data Mover: 1-D tile, 2048 x 2-byte elements -> Vt ---
      unsigned long long ga = (unsigned long long)(uintptr_t)(vp + (size_t)kc * HD);
      u32x4 g0;
      g0[0] = 1u;                                             // count=1 (valid D#)
      g0[1] = ldsV;                                           // lds_addr
      g0[2] = (unsigned)(ga & 0xFFFFFFFFu);                   // global_addr lo
      g0[3] = (unsigned)((ga >> 32) & 0x01FFFFFFu) | (2u << 30); // addr hi | type=2
      i32x8 g1;
      g1[0] = 1 << 16;               // data_size = 1 (2 bytes)
      g1[1] = (int)(2048u << 16);    // tensor_dim0 = 2048 (bits [79:48] lo half)
      g1[2] = (int)(1u << 16);       // tensor_dim0 hi = 0 | tensor_dim1 = 1
      g1[3] = (int)(2048u << 16);    // tensor_dim1 hi = 0 | tile_dim0 = 2048
      g1[4] = 0;                     // tile_dim1 = 0 (unused), tile_dim2 = 0
      g1[5] = 2048;                  // tensor_dim0_stride lo
      g1[6] = 0; g1[7] = 0;
      i32x4 z4 = {0, 0, 0, 0};
      i32x8 z8 = {0, 0, 0, 0, 0, 0, 0, 0};
      __builtin_amdgcn_tensor_load_to_lds(g0, g1, z4, z4, z8, 0);
    }
    { // all threads: K chunk transposed into Kt (2048 halfs, 8 per thread)
      int key = tid >> 3, c8 = (tid & 7) * 8;
      float4 kv = *(const float4*)(kp + (size_t)(kc + key) * HD + c8);
      const _Float16* ks = (const _Float16*)&kv;
#pragma unroll
      for (int i = 0; i < 8; ++i) Kt[c8 + i][key] = ks[i];
    }
    if (wave == 0) __builtin_amdgcn_s_wait_tensorcnt(0);
    __syncthreads();

    // S(16x32) = Q(16x64) @ K^T : 4 WMMAs (2 key tiles x 2 d-chunks)
    v8f S0 = {}, S1 = {};
    {
      v16h kb;
#pragma unroll
      for (int i = 0; i < 16; ++i) kb[i] = Kt[(i < 8) ? (hh * 8 + i) : (8 + hh * 8 + i)][lh];
      S0 = wmma_f16(qa0, kb, S0);
#pragma unroll
      for (int i = 0; i < 16; ++i) kb[i] = Kt[((i < 8) ? (hh * 8 + i) : (8 + hh * 8 + i)) + 32][lh];
      S0 = wmma_f16(qa1, kb, S0);
#pragma unroll
      for (int i = 0; i < 16; ++i) kb[i] = Kt[(i < 8) ? (hh * 8 + i) : (8 + hh * 8 + i)][16 + lh];
      S1 = wmma_f16(qa0, kb, S1);
#pragma unroll
      for (int i = 0; i < 16; ++i) kb[i] = Kt[((i < 8) ? (hh * 8 + i) : (8 + hh * 8 + i)) + 32][16 + lh];
      S1 = wmma_f16(qa1, kb, S1);
    }

    // online softmax over C layout: row r+8*hh lives on 16 lanes of the half
#pragma unroll
    for (int r = 0; r < 8; ++r) {
      float vmax = fmaxf(S0[r], S1[r]);
#pragma unroll
      for (int msk = 1; msk < 16; msk <<= 1) vmax = fmaxf(vmax, __shfl_xor(vmax, msk));
      float mnew = fmaxf(mrow[r], vmax);
      float scl = __expf(mrow[r] - mnew);
      float p0 = __expf(S0[r] - mnew);
      float p1 = __expf(S1[r] - mnew);
      float psum = p0 + p1;
#pragma unroll
      for (int msk = 1; msk < 16; msk <<= 1) psum += __shfl_xor(psum, msk);
      lrow[r] = lrow[r] * scl + psum;
      mrow[r] = mnew;
      O0[r] *= scl; O1[r] *= scl; O2[r] *= scl; O3[r] *= scl;
      Ps[wave][r + 8 * hh][lh]      = (_Float16)p0;
      Ps[wave][r + 8 * hh][lh + 16] = (_Float16)p1;
    }

    // relayout P (C layout -> A layout) through per-wave LDS
    v16h pa;
#pragma unroll
    for (int p = 0; p < 8; ++p) {
      int kk = (p < 4) ? (hh * 8 + 2 * p) : (16 + hh * 8 + 2 * (p - 4));
      pa[2 * p]     = Ps[wave][lh][kk];
      pa[2 * p + 1] = Ps[wave][lh][kk + 1];
    }
    // O(16x64) += P(16x32) @ V(32x64): 4 WMMAs
    v16h vb;
#pragma unroll
    for (int i = 0; i < 16; ++i) vb[i] = Vt[(i < 8) ? (hh * 8 + i) : (8 + hh * 8 + i)][lh];
    O0 = wmma_f16(pa, vb, O0);
#pragma unroll
    for (int i = 0; i < 16; ++i) vb[i] = Vt[(i < 8) ? (hh * 8 + i) : (8 + hh * 8 + i)][16 + lh];
    O1 = wmma_f16(pa, vb, O1);
#pragma unroll
    for (int i = 0; i < 16; ++i) vb[i] = Vt[(i < 8) ? (hh * 8 + i) : (8 + hh * 8 + i)][32 + lh];
    O2 = wmma_f16(pa, vb, O2);
#pragma unroll
    for (int i = 0; i < 16; ++i) vb[i] = Vt[(i < 8) ? (hh * 8 + i) : (8 + hh * 8 + i)][48 + lh];
    O3 = wmma_f16(pa, vb, O3);
    __syncthreads();
  }

  // normalize and store f16 [token, EMBED] as next GEMM's A matrix
#pragma unroll
  for (int r = 0; r < 8; ++r) {
    int n = qrow + r + 8 * hh;
    float inv = 1.0f / lrow[r];
    size_t base = ((size_t)(b * NMAX + n)) * EMBED + h * HD;
    hO[base + 0  + lh] = (_Float16)(O0[r] * inv);
    hO[base + 16 + lh] = (_Float16)(O1[r] * inv);
    hO[base + 32 + lh] = (_Float16)(O2[r] * inv);
    hO[base + 48 + lh] = (_Float16)(O3[r] * inv);
  }
}

__global__ void mask_kernel(float* __restrict__ out, int4 lens) {
  int i = blockIdx.x * 256 + threadIdx.x;
  if (i >= BATCH * NMAX) return;
  int b = i / NMAX, n = i % NMAX;
  int len = (b == 0) ? lens.x : (b == 1) ? lens.y : (b == 2) ? lens.z : lens.w;
  out[i] = (n >= len) ? 1.0f : 0.0f;
}

// ---------------------------------------------------------------------------
extern "C" void kernel_launch(void* const* d_in, const int* in_sizes, int n_in,
                              void* d_out, int out_size, void* d_ws, size_t ws_size,
                              hipStream_t stream) {
  const float* img0 = (const float*)d_in[0];
  const float* img1 = (const float*)d_in[1];
  const float* img2 = (const float*)d_in[2];
  const float* img3 = (const float*)d_in[3];
  const float* pe_ln_in_w  = (const float*)d_in[4];
  const float* pe_ln_in_b  = (const float*)d_in[5];
  const float* pe_w        = (const float*)d_in[6];
  const float* pe_ln_out_w = (const float*)d_in[7];
  const float* pe_ln_out_b = (const float*)d_in[8];
  const float* ln1_w  = (const float*)d_in[9];
  const float* ln1_b  = (const float*)d_in[10];
  const float* qkv_w  = (const float*)d_in[11];
  const float* qkv_b  = (const float*)d_in[12];
  const float* proj_w = (const float*)d_in[13];
  const float* proj_b = (const float*)d_in[14];
  const float* ln2_w  = (const float*)d_in[15];
  const float* ln2_b  = (const float*)d_in[16];
  const float* fc1_w  = (const float*)d_in[17];
  const float* fc1_b  = (const float*)d_in[18];
  const float* fc2_w  = (const float*)d_in[19];
  const float* fc2_b  = (const float*)d_in[20];
  const float* lnf_w  = (const float*)d_in[21];
  const float* lnf_b  = (const float*)d_in[22];

  // ---- workspace carve (deterministic, ~62 MB) ----
  char* wp = (char*)d_ws;
  auto carve = [&](size_t bytes) -> void* {
    void* p = (void*)wp;
    wp += (bytes + 255) & ~(size_t)255;
    return p;
  };
  _Float16* hA96 = (_Float16*)carve((size_t)MROWS * PD * 2);
  _Float16* hLN  = (_Float16*)carve((size_t)MROWS * EMBED * 2);
  _Float16* hO   = (_Float16*)carve((size_t)MROWS * EMBED * 2);
  _Float16* hFF  = (_Float16*)carve((size_t)MROWS * FF * 2);
  float*    xbuf = (float*)   carve((size_t)MROWS * EMBED * 4);
  float*    tmpb = (float*)   carve((size_t)MROWS * EMBED * 4);
  float*    qkvb = (float*)   carve((size_t)MROWS * QKVC * 4);
  _Float16* qf   = (_Float16*)carve((size_t)MROWS * EMBED * 2);  // B*H*N*HD
  _Float16* kf   = (_Float16*)carve((size_t)MROWS * EMBED * 2);
  _Float16* vf   = (_Float16*)carve((size_t)MROWS * EMBED * 2);
  float*    cosb = (float*)   carve((size_t)MROWS * 32 * 4);
  float*    sinb = (float*)   carve((size_t)MROWS * 32 * 4);

  int4 lens; lens.x = 512; lens.y = 768; lens.z = 896; lens.w = 1024;

  // patch embed: LN(96) -> GEMM(96->384) -> LN -> x
  prep_kernel<<<dim3(NMAX, BATCH), 128, 0, stream>>>(
      img0, img1, img2, img3, pe_ln_in_w, pe_ln_in_b, hA96, cosb, sinb);
  gemm_kernel<0><<<dim3(EMBED / 128, MROWS / 64), 256, 0, stream>>>(
      hA96, pe_w, nullptr, tmpb, nullptr, EMBED, PD);
  ln_kernel<<<MROWS, 128, 0, stream>>>(tmpb, pe_ln_out_w, pe_ln_out_b, nullptr, xbuf);

  for (int l = 0; l < DEPTH; ++l) {
    ln_kernel<<<MROWS, 128, 0, stream>>>(xbuf, ln1_w + l * EMBED, ln1_b + l * EMBED, hLN, nullptr);
    gemm_kernel<0><<<dim3(QKVC / 128, MROWS / 64), 256, 0, stream>>>(
        hLN, qkv_w + (size_t)l * EMBED * QKVC, qkv_b + l * QKVC, qkvb, nullptr, QKVC, EMBED);
    rope_kernel<<<dim3(NMAX, BATCH), 128, 0, stream>>>(qkvb, cosb, sinb, qf, kf, vf);
    attn_kernel<<<dim3(NMAX / 128, HEADS, BATCH), 256, 0, stream>>>(qf, kf, vf, hO, lens);
    gemm_kernel<1><<<dim3(EMBED / 128, MROWS / 64), 256, 0, stream>>>(
        hO, proj_w + (size_t)l * EMBED * EMBED, proj_b + l * EMBED, xbuf, nullptr, EMBED, EMBED);
    ln_kernel<<<MROWS, 128, 0, stream>>>(xbuf, ln2_w + l * EMBED, ln2_b + l * EMBED, hLN, nullptr);
    gemm_kernel<2><<<dim3(FF / 128, MROWS / 64), 256, 0, stream>>>(
        hLN, fc1_w + (size_t)l * EMBED * FF, fc1_b + l * FF, nullptr, hFF, FF, EMBED);
    gemm_kernel<1><<<dim3(EMBED / 128, MROWS / 64), 256, 0, stream>>>(
        hFF, fc2_w + (size_t)l * FF * EMBED, fc2_b + l * EMBED, xbuf, nullptr, EMBED, FF);
  }

  ln_kernel<<<MROWS, 128, 0, stream>>>(xbuf, lnf_w, lnf_b, nullptr, (float*)d_out);
  mask_kernel<<<(BATCH * NMAX + 255) / 256, 256, 0, stream>>>(
      (float*)d_out + (size_t)MROWS * EMBED, lens);
}
